// MultiMemoryAttention_15401752724099
// MI455X (gfx1250) — compile-verified, matrix operand order checked by reading, not run on hardware
//
#include <hip/hip_runtime.h>
#include <hip/hip_bf16.h>

typedef __attribute__((ext_vector_type(16))) _Float16 v16h;
typedef __attribute__((ext_vector_type(8)))  _Float16 v8h;
typedef __attribute__((ext_vector_type(8)))  float    v8f;

#define DI __device__ __forceinline__

// ---------------------------------------------------------------------------
// WMMA helpers (gfx1250: wave32, 16x16x32 f16 -> f32)
// ---------------------------------------------------------------------------
DI v8f wmma_f16(v16h a, v16h b, v8f c) {
    return __builtin_amdgcn_wmma_f32_16x16x32_f16(
        /*neg_a=*/false, a, /*neg_b=*/false, b,
        /*c_mod=*/(short)0, c, /*reuse_a=*/false, /*reuse_b=*/false);
}

// Load a 16x32 f16 A/B fragment from a row-major [rows, ld] f16 source.
// Works for both A (rows = M) and B (rows = N of B^T) because every product
// here is written as A_rm[M,K] @ (B_rm[N,K])^T.
// Layout per ISA 7.12.2 (16-bit A 16x32):
//   lanes 0-15 : halves 0..7 = K k0..k0+7,  halves 8..15 = K k0+16..k0+23
//   lanes16-31 : halves 0..7 = K k0+8..k0+15, halves 8..15 = K k0+24..k0+31
DI v16h load_frag(const _Float16* base, int ld, int rowbase, int k0) {
    const int lane = threadIdx.x & 31;
    const int row  = rowbase + (lane & 15);
    const _Float16* p = base + (size_t)row * ld + k0 + ((lane & 16) ? 8 : 0);
    v8h lo = *(const v8h*)(p);
    v8h hi = *(const v8h*)(p + 16);
    v16h r;
#pragma unroll
    for (int i = 0; i < 8; ++i) { r[i] = lo[i]; r[8 + i] = hi[i]; }
    return r;
}

DI float elu1f(float x) {
    return x > 0.f ? x + 1.f : __expf(fminf(x, 0.f));
}

// ---------------------------------------------------------------------------
// Elementwise f32 -> f16 convert
// ---------------------------------------------------------------------------
__global__ void cvt_f32_f16_kernel(const float* __restrict__ in,
                                   _Float16* __restrict__ out, int n) {
    int i = blockIdx.x * 256 + threadIdx.x;
    if (i < n) out[i] = (_Float16)in[i];
}

// Transpose memories [M*nh, 128, 128] f32 -> memT [M*nh, e, d] f16
__global__ void transpose_mem_kernel(const float* __restrict__ memories,
                                     _Float16* __restrict__ memT) {
    const int mh = blockIdx.x;                   // 0..63
    const float* src = memories + (size_t)mh * 128 * 128;
    _Float16*    dst = memT     + (size_t)mh * 128 * 128;
    for (int i = threadIdx.x; i < 128 * 128; i += 256) {
        int d = i >> 7, e = i & 127;
        dst[(size_t)e * 128 + d] = (_Float16)src[(size_t)d * 128 + e];
    }
}

// ---------------------------------------------------------------------------
// GEMM: out[i,j] = sum_k A16[i,k] * W16[j,k]   (i.e. A @ W^T)
// A16: [8192, 2048] f16, W: [2048, 2048] f16 (row-major)
// final_mode==0: blockIdx.z selects wq/wk/wv; epilogue scatters to
//   [B, nh, S, hd] with elu+1 applied for q/k (f16) and raw f32 q kept.
// final_mode==1: plain row-major f32 store to outf.
// Block: 256 thr = 8 waves; block tile 128x128; wave tile 64(M) x 32(N).
// ---------------------------------------------------------------------------
__global__ void gemm_wmma_kernel(const _Float16* __restrict__ A16,
                                 const _Float16* __restrict__ Wall,
                                 float* __restrict__ qf32,
                                 _Float16* __restrict__ sq16,
                                 _Float16* __restrict__ sk16,
                                 _Float16* __restrict__ v16p,
                                 float* __restrict__ outf,
                                 int final_mode) {
    const int K = 2048;
    const int z = blockIdx.z;
    const _Float16* W = Wall + (size_t)z * K * K;

    const int wid = threadIdx.x >> 5;
    const int wm  = wid & 1;        // 2 M-strips of 64
    const int wn  = wid >> 1;       // 4 N-strips of 32
    const int Mb  = blockIdx.x * 128 + wm * 64;
    const int Nb  = blockIdx.y * 128 + wn * 32;
    const int lane = threadIdx.x & 31;

    v8f acc[4][2];
#pragma unroll
    for (int i = 0; i < 4; ++i)
#pragma unroll
        for (int j = 0; j < 2; ++j)
#pragma unroll
            for (int r = 0; r < 8; ++r) acc[i][j][r] = 0.f;

    for (int kk = 0; kk < K; kk += 32) {
        // software prefetch two K-steps ahead (global_prefetch_b8 on gfx1250)
        __builtin_prefetch(A16 + (size_t)(Mb + (lane & 15)) * K + kk + 64, 0, 1);
        __builtin_prefetch(W   + (size_t)(Nb + (lane & 15)) * K + kk + 64, 0, 1);

        v16h a[4], b[2];
#pragma unroll
        for (int i = 0; i < 4; ++i) a[i] = load_frag(A16, K, Mb + i * 16, kk);
#pragma unroll
        for (int j = 0; j < 2; ++j) b[j] = load_frag(W, K, Nb + j * 16, kk);
#pragma unroll
        for (int i = 0; i < 4; ++i)
#pragma unroll
            for (int j = 0; j < 2; ++j)
                acc[i][j] = wmma_f16(a[i], b[j], acc[i][j]);
    }

    // Epilogue. C/D layout: lane 0-15 -> N=lane, M=vgpr r; lane 16-31 -> M=r+8.
    const int nlo  = lane & 15;
    const int mhi  = (lane & 16) ? 8 : 0;
#pragma unroll
    for (int i = 0; i < 4; ++i) {
#pragma unroll
        for (int j = 0; j < 2; ++j) {
            const int n = Nb + j * 16 + nlo;
#pragma unroll
            for (int r = 0; r < 8; ++r) {
                const int m = Mb + i * 16 + mhi + r;
                const float val = acc[i][j][r];
                if (final_mode) {
                    outf[(size_t)m * 2048 + n] = val;
                } else {
                    // m = b*4096 + s ; n = h*128 + d  -> [B, nh, S, hd]
                    const int b_ = m >> 12, s = m & 4095;
                    const int h  = n >> 7,  d = n & 127;
                    const size_t idx =
                        (((size_t)(b_ * 16 + h)) * 4096 + s) * 128 + d;
                    if (z == 0) {
                        qf32[idx] = val;
                        sq16[idx] = (_Float16)elu1f(val);
                    } else if (z == 1) {
                        sk16[idx] = (_Float16)elu1f(val);
                    } else {
                        v16p[idx] = (_Float16)val;
                    }
                }
            }
        }
    }
}

// ---------------------------------------------------------------------------
// rel[m,b,h] = mean_s ( q[b,h,s,:] . landmarks[m,h,:] )
// one block per (m, bh); 256 threads reduce over S=4096
// ---------------------------------------------------------------------------
__global__ void rel_kernel(const float* __restrict__ qf32,
                           const float* __restrict__ landmarks,
                           float* __restrict__ rel) {
    __shared__ float red[256];
    const int m  = blockIdx.x >> 5;
    const int bh = blockIdx.x & 31;
    const int h  = bh & 15;
    const float* lm = landmarks + (size_t)(m * 16 + h) * 128;
    const float* qB = qf32 + (size_t)bh * 4096 * 128;
    float p = 0.f;
    for (int s = threadIdx.x; s < 4096; s += 256) {
        const float* row = qB + (size_t)s * 128;
        float d = 0.f;
        for (int i = 0; i < 128; ++i) d += row[i] * lm[i];
        p += d;
    }
    red[threadIdx.x] = p;
    __syncthreads();
    for (int st = 128; st > 0; st >>= 1) {
        if (threadIdx.x < st) red[threadIdx.x] += red[threadIdx.x + st];
        __syncthreads();
    }
    if (threadIdx.x == 0) rel[blockIdx.x] = red[0] * (1.f / 4096.f);
}

// softmax over M=4 memories per (b,h), with key_counts active mask
__global__ void softw_kernel(const float* __restrict__ rel,
                             const float* __restrict__ key_counts,
                             float* __restrict__ wgt) {
    const int bh = threadIdx.x;
    if (bh >= 32) return;
    float r[4];
    for (int m = 0; m < 4; ++m) {
        float kc = 0.f;
        for (int h2 = 0; h2 < 16; ++h2) kc += key_counts[m * 16 + h2];
        r[m] = (kc >= 1e-6f) ? rel[m * 32 + bh] : -1e30f;
    }
    float mx = r[0];
    for (int m = 1; m < 4; ++m) mx = fmaxf(mx, r[m]);
    float e[4], se = 0.f;
    for (int m = 0; m < 4; ++m) { e[m] = __expf(r[m] - mx); se += e[m]; }
    for (int m = 0; m < 4; ++m)
        wgt[m * 32 + bh] = (se > 0.f) ? e[m] / se : 0.f;
}

// ---------------------------------------------------------------------------
// Memory retrieval: mix[b,s,h*128+e] = g_h * sum_m w_m * (sq@mem_m)[s,e]/norm_m[s]
// grid (bh=32, s-tiles=32), 256 thr = 8 waves, each wave a 16-row strip.
// A-fragments of sq are hoisted: loaded once, reused across all 8x4 tiles.
// ---------------------------------------------------------------------------
__global__ void retrieve_kernel(const _Float16* __restrict__ sq16,
                                const _Float16* __restrict__ memT,
                                const float* __restrict__ mnorms,
                                const float* __restrict__ wgt,
                                const float* __restrict__ gate,
                                float* __restrict__ mix) {
    __shared__ float normv[4][128];
    __shared__ float wz[4];
    const int bh = blockIdx.x, h = bh & 15, b_ = bh >> 4;
    const int s0 = blockIdx.y * 128;
    const int tid = threadIdx.x, lane = tid & 31, wid = tid >> 5;
    const _Float16* sqB = sq16 + ((size_t)bh * 4096 + s0) * 128;

    if (tid < 4) wz[tid] = wgt[tid * 32 + bh];
    if (tid < 128) {
        const _Float16* row = sqB + (size_t)tid * 128;
        for (int m = 0; m < 4; ++m) {
            const float* nr = mnorms + (size_t)(m * 16 + h) * 128;
            float s = 0.f;
            for (int d = 0; d < 128; ++d) s += (float)row[d] * nr[d];
            normv[m][tid] = fmaxf(s, 1e-6f);
        }
    }
    __syncthreads();

    const float g = 1.f / (1.f + __expf(-gate[h]));
    const int Mb = wid * 16;
    const int nlo = lane & 15, mhi = (lane & 16) ? 8 : 0;

    v16h aq[4];
#pragma unroll
    for (int kk = 0; kk < 4; ++kk) aq[kk] = load_frag(sqB, 128, Mb, kk * 32);

    for (int jf = 0; jf < 8; ++jf) {
        float outv[8];
#pragma unroll
        for (int r = 0; r < 8; ++r) outv[r] = 0.f;
        for (int m = 0; m < 4; ++m) {
            const _Float16* BT = memT + (size_t)(m * 16 + h) * 128 * 128; // [e][d]
            v8f acc;
#pragma unroll
            for (int r = 0; r < 8; ++r) acc[r] = 0.f;
#pragma unroll
            for (int kk = 0; kk < 4; ++kk)
                acc = wmma_f16(aq[kk], load_frag(BT, 128, jf * 16, kk * 32), acc);
#pragma unroll
            for (int r = 0; r < 8; ++r) {
                const int q = Mb + mhi + r;
                outv[r] += wz[m] * acc[r] / normv[m][q];
            }
        }
        const int e = jf * 16 + nlo;
#pragma unroll
        for (int r = 0; r < 8; ++r) {
            const int q = Mb + mhi + r;
            const size_t oi =
                ((size_t)(b_ * 4096 + s0 + q)) * 2048 + h * 128 + e;
            mix[oi] = g * outv[r];
        }
    }
}

// ---------------------------------------------------------------------------
// Chunked causal linear attention; one block per (b,h), sequential over 32
// chunks of 128. Running state S^T[e,d] and z[d] live in LDS.
// Dynamic LDS: 4 x (128*128 f16) + 2*128 f32 = 132096 B (within 320KB WGP LDS)
// ---------------------------------------------------------------------------
__global__ void linattn_kernel(const _Float16* __restrict__ sq16,
                               const _Float16* __restrict__ sk16,
                               const _Float16* __restrict__ v16p,
                               const float* __restrict__ gate,
                               float* __restrict__ mix) {
    extern __shared__ char smem_raw[];
    _Float16* sT  = (_Float16*)smem_raw;   // [e][d]  running state (transposed)
    _Float16* vT  = sT  + 16384;           // [e][kc] chunk of v, transposed
    _Float16* skT = vT  + 16384;           // [d][kc] chunk of sk, transposed
    _Float16* scr = skT + 16384;           // [q][kc] masked scores (f16)
    float* zb  = (float*)(scr + 16384);    // [128]   running z
    float* nrm = zb + 128;                 // [128]   per-row norm

    const int bh = blockIdx.x, h = bh & 15, b_ = bh >> 4;
    const int tid = threadIdx.x, lane = tid & 31, wid = tid >> 5;
    const float g  = 1.f / (1.f + __expf(-gate[h]));
    const float om = 1.f - g;

    const _Float16* sqB = sq16 + (size_t)bh * 4096 * 128;
    const _Float16* skB = sk16 + (size_t)bh * 4096 * 128;
    const _Float16* vB  = v16p + (size_t)bh * 4096 * 128;

    for (int i = tid; i < 16384; i += 256) sT[i] = (_Float16)0.f;
    if (tid < 128) zb[tid] = 0.f;
    __syncthreads();

    const int nlo = lane & 15, mhi = (lane & 16) ? 8 : 0;
    const int Mb = wid * 16;     // wave's 16-row strip (q rows / e rows)

    for (int c = 0; c < 32; ++c) {
        const int s0 = c * 128;
        // stage transposed chunk tiles
        for (int i = tid; i < 16384; i += 256) {
            const int kc = i >> 7, e = i & 127;
            vT[e * 128 + kc]  = vB[(size_t)(s0 + kc) * 128 + e];
            skT[e * 128 + kc] = skB[(size_t)(s0 + kc) * 128 + e];
        }
        __syncthreads();

        // sq A-fragments for this chunk: used by scores stage AND out stage
        v16h aq[4];
#pragma unroll
        for (int kk = 0; kk < 4; ++kk)
            aq[kk] = load_frag(sqB + (size_t)s0 * 128, 128, Mb, kk * 32);

        // scores = sq @ sk^T, masked causal, -> scr (f16)
#pragma unroll
        for (int jf = 0; jf < 8; ++jf) {
            v8f acc;
#pragma unroll
            for (int r = 0; r < 8; ++r) acc[r] = 0.f;
#pragma unroll
            for (int kk = 0; kk < 4; ++kk)
                acc = wmma_f16(aq[kk],
                               load_frag(skB + (size_t)s0 * 128, 128,
                                         jf * 16, kk * 32),
                               acc);
            const int n = jf * 16 + nlo;
#pragma unroll
            for (int r = 0; r < 8; ++r) {
                const int m = Mb + mhi + r;
                scr[m * 128 + n] = (_Float16)((n <= m) ? acc[r] : 0.f);
            }
        }
        __syncthreads();

        // norm[q] = max( rowsum(scores) + sq[q,:].z , eps )   (z pre-update)
        if (tid < 128) {
            const int q = tid;
            float rsum = 0.f;
            for (int kc = 0; kc < 128; ++kc) rsum += (float)scr[q * 128 + kc];
            float nz = 0.f;
            const _Float16* qq = sqB + (size_t)(s0 + q) * 128;
            for (int d = 0; d < 128; ++d) nz += (float)qq[d] * zb[d];
            nrm[q] = fmaxf(rsum + nz, 1e-6f);
        }
        __syncthreads();

        // out = (sq @ S + scores @ v) / norm ; accumulate (1-g)*out into mix
        {
            v16h ascr[4];
#pragma unroll
            for (int kk = 0; kk < 4; ++kk)
                ascr[kk] = load_frag(scr, 128, Mb, kk * 32);
#pragma unroll
            for (int jf = 0; jf < 8; ++jf) {
                v8f acc;
#pragma unroll
                for (int r = 0; r < 8; ++r) acc[r] = 0.f;
#pragma unroll
                for (int kk = 0; kk < 4; ++kk)
                    acc = wmma_f16(aq[kk],
                                   load_frag(sT, 128, jf * 16, kk * 32), acc);
#pragma unroll
                for (int kk = 0; kk < 4; ++kk)
                    acc = wmma_f16(ascr[kk],
                                   load_frag(vT, 128, jf * 16, kk * 32), acc);
                const int e = jf * 16 + nlo;
#pragma unroll
                for (int r = 0; r < 8; ++r) {
                    const int q = Mb + mhi + r;
                    const float val = acc[r] / nrm[q];
                    const size_t oi =
                        ((size_t)(b_ * 4096 + s0 + q)) * 2048 + h * 128 + e;
                    mix[oi] += om * val;
                }
            }
        }

        // U[e,d] = v^T @ sk  (chunk contribution to S^T), then S^T += U
        v8f U[8];
        {
            v16h av[4];
#pragma unroll
            for (int kk = 0; kk < 4; ++kk)
                av[kk] = load_frag(vT, 128, Mb, kk * 32);
#pragma unroll
            for (int jf = 0; jf < 8; ++jf) {
                v8f acc;
#pragma unroll
                for (int r = 0; r < 8; ++r) acc[r] = 0.f;
#pragma unroll
                for (int kk = 0; kk < 4; ++kk)
                    acc = wmma_f16(av[kk],
                                   load_frag(skT, 128, jf * 16, kk * 32), acc);
                U[jf] = acc;
            }
        }
        __syncthreads();   // all reads of sT / zb for this chunk are done
#pragma unroll
        for (int jf = 0; jf < 8; ++jf) {
            const int d = jf * 16 + nlo;
#pragma unroll
            for (int r = 0; r < 8; ++r) {
                const int e = Mb + mhi + r;
                sT[e * 128 + d] =
                    (_Float16)((float)sT[e * 128 + d] + U[jf][r]);
            }
        }
        if (tid < 128) {
            const int d = tid;
            float s = 0.f;
            for (int kc = 0; kc < 128; ++kc) s += (float)skT[d * 128 + kc];
            zb[d] += s;
        }
        __syncthreads();
    }
}

// ---------------------------------------------------------------------------
// Host launcher
// ---------------------------------------------------------------------------
extern "C" void kernel_launch(void* const* d_in, const int* in_sizes, int n_in,
                              void* d_out, int out_size, void* d_ws, size_t ws_size,
                              hipStream_t stream) {
    (void)in_sizes; (void)n_in; (void)out_size; (void)ws_size;
    const float* hidden     = (const float*)d_in[0];
    const float* wq         = (const float*)d_in[1];
    const float* wk         = (const float*)d_in[2];
    const float* wv         = (const float*)d_in[3];
    const float* wo         = (const float*)d_in[4];
    const float* gate       = (const float*)d_in[5];
    const float* memories   = (const float*)d_in[6];
    const float* mem_norms  = (const float*)d_in[7];
    const float* landmarks  = (const float*)d_in[8];
    const float* key_counts = (const float*)d_in[9];
    float* out = (float*)d_out;

    const size_t BS = 8192, Hdim = 2048, NE = BS * Hdim; // 16,777,216
    char* ws = (char*)d_ws;
    size_t o = 0;
    _Float16* h16  = (_Float16*)(ws + o); o += NE * 2;                 // hidden f16 / mix16
    _Float16* w16  = (_Float16*)(ws + o); o += 4 * Hdim * Hdim * 2;    // wq,wk,wv,wo f16
    _Float16* memT = (_Float16*)(ws + o); o += (size_t)64 * 128 * 128 * 2;
    float*    qf32 = (float*)(ws + o);    o += NE * 4;                 // raw q [B,nh,S,hd]
    _Float16* sq16 = (_Float16*)(ws + o); o += NE * 2;
    _Float16* sk16 = (_Float16*)(ws + o); o += NE * 2;
    _Float16* v16p = (_Float16*)(ws + o); o += NE * 2;
    float*    mix  = (float*)(ws + o);    o += NE * 4;                 // [B,S,H]
    float*    rel  = (float*)(ws + o);    o += 512;
    float*    wgt  = (float*)(ws + o);    o += 512;

    // 1) f32 -> f16 conversions
    cvt_f32_f16_kernel<<<(int)(NE / 256), 256, 0, stream>>>(hidden, h16, (int)NE);
    const int WN = (int)(Hdim * Hdim);
    cvt_f32_f16_kernel<<<WN / 256, 256, 0, stream>>>(wq, w16 + 0 * (size_t)WN, WN);
    cvt_f32_f16_kernel<<<WN / 256, 256, 0, stream>>>(wk, w16 + 1 * (size_t)WN, WN);
    cvt_f32_f16_kernel<<<WN / 256, 256, 0, stream>>>(wv, w16 + 2 * (size_t)WN, WN);
    cvt_f32_f16_kernel<<<WN / 256, 256, 0, stream>>>(wo, w16 + 3 * (size_t)WN, WN);
    transpose_mem_kernel<<<64, 256, 0, stream>>>(memories, memT);

    // 2) fused QKV projection GEMMs (z = 0/1/2), elu+1 epilogue for q/k
    gemm_wmma_kernel<<<dim3(64, 16, 3), 256, 0, stream>>>(
        h16, w16, qf32, sq16, sk16, v16p, nullptr, 0);

    // 3) landmark relevance + softmax memory weights
    rel_kernel<<<128, 256, 0, stream>>>(qf32, landmarks, rel);
    softw_kernel<<<1, 32, 0, stream>>>(rel, key_counts, wgt);

    // 4) memory retrieval -> mix = g * mem_out   (writes every element)
    retrieve_kernel<<<dim3(32, 32), 256, 0, stream>>>(
        sq16, memT, mem_norms, wgt, gate, mix);

    // 5) causal linear attention -> mix += (1-g) * local
    linattn_kernel<<<32, 256, 132096, stream>>>(sq16, sk16, v16p, gate, mix);

    // 6) final projection: out = mix @ wo^T
    cvt_f32_f16_kernel<<<(int)(NE / 256), 256, 0, stream>>>(mix, h16, (int)NE);
    gemm_wmma_kernel<<<dim3(64, 16, 1), 256, 0, stream>>>(
        h16, w16 + 3 * (size_t)WN, nullptr, nullptr, nullptr, nullptr, out, 1);
}
